// RelationalMemoryAugv3_35940286333501
// MI455X (gfx1250) — matrix operand order repaired and checked by reading, not compiled, not run on hardware
//
#include <hip/hip_runtime.h>
#include <math.h>

#define NSLOT 3
#define DM    512
#define DKK   64
#define BB    256
#define TT    128
#define NB    4              // batch elements per workgroup
#define NWG   (BB / NB)      // 64 workgroups
#define ROWS  16             // NB * (NSLOT+1) kv rows -> exactly one WMMA M tile
#define STR   520            // LDS row stride (elems), 16B aligned, bank-staggered
#define STRG  1040           // LDS row stride for 1024-wide buffers

// ---- workspace layout (bf16 elems unless noted) ----
#define OFF_WQKV 0
#define OFF_WO   786432
#define OFF_WM1  1048576
#define OFF_WM2  1310720
#define OFF_WG   1572864
#define OFF_HEAD 2621440                  // 5 x [64][64]
#define ELEMS_BF 2641920
#define OFF_BQKV_BYTES (ELEMS_BF * 2)     // f32[1536]
#define OFF_BG_BYTES   (OFF_BQKV_BYTES + 1536 * 4) // f32[1024]
#define PREP_ELEMS (ELEMS_BF + 1536 + 1024)

#define SMEM_BYTES 269312

typedef __attribute__((ext_vector_type(16))) __bf16 v16bf;
typedef __attribute__((ext_vector_type(8)))  float  v8f;

union PackBF { struct { uint4 lo, hi; } q; v16bf v; };

__device__ __forceinline__ float bf2f(unsigned short h) {
  union { unsigned int u; float f; } c; c.u = ((unsigned int)h) << 16; return c.f;
}
__device__ __forceinline__ unsigned short f2bf(float f) {
  union { float f; unsigned int u; } c; c.f = f;
  unsigned int u = c.u;
  return (unsigned short)((u + 0x7FFFu + ((u >> 16) & 1u)) >> 16);
}
__device__ __forceinline__ float sigmoidf_(float x) { return 1.0f / (1.0f + __expf(-x)); }

#define EPI_BF16    0
#define EPI_NM      1
#define EPI_ADDRELU 2
#define EPI_F32     3

// C[16][N] = A[16][K] @ W[K][N] (+bias, epilogue). A: LDS bf16. Wt: global bf16 [N or wt_mod][K].
// wt_mod (power of 2): 64 for per-head block-diagonal weights, else N.
// Processes TWO N-tiles per wave iteration -> two independent WMMA accumulator
// chains (hides XDL latency). SHARED_A=true when both tiles read the same A
// fragment (wt_mod == N), halving LDS A traffic.
template <bool SHARED_A>
__device__ __forceinline__ void gemm16(
    const unsigned short* __restrict__ A, int a_stride,
    const unsigned short* __restrict__ Wt, int wt_stride,
    const float* __restrict__ bias,
    int N, int K, int wt_mod, int epi, int do_relu,
    unsigned short* __restrict__ dstb, float* __restrict__ dstf, int d_stride,
    const float* __restrict__ addsrc, int add_stride, int tid)
{
  const int wave = tid >> 5;
  const int lane = tid & 31;
  const int g    = lane >> 4;   // K-half group
  const int l16  = lane & 15;   // A row / B (C) column within tile
  const int ntiles = N >> 4;    // always a multiple of 16 here (32 or 64)

  auto epi_one = [&](const v8f& acc, int ncol, float bsv) {
#pragma unroll
    for (int r = 0; r < 8; ++r) {
      const int row = r + 8 * g;
      float v = acc[r] + bsv;
      if (do_relu) v = fmaxf(v, 0.0f);
      if (epi == EPI_BF16)         dstb[row * d_stride + ncol] = f2bf(v);
      else if (epi == EPI_NM)      dstf[row * d_stride + ncol] = addsrc[row * add_stride + ncol] + v;
      else if (epi == EPI_ADDRELU) dstf[row * d_stride + ncol] += v;
      else                         dstf[row * d_stride + ncol] = v;
    }
  };

  for (int nt = wave; nt < ntiles; nt += 16) {
    const int ncol0 = (nt << 4) + l16;
    const int ncol1 = ncol0 + 128;               // tile nt+8
    const int wrow0 = ncol0 & (wt_mod - 1);
    const int wrow1 = ncol1 & (wt_mod - 1);
    const unsigned short* ap0 = A + l16 * a_stride + (ncol0 - wrow0) + 8 * g;
    const unsigned short* ap1 = A + l16 * a_stride + (ncol1 - wrow1) + 8 * g;
    const unsigned short* bp0 = Wt + (size_t)wrow0 * wt_stride + 8 * g;
    const unsigned short* bp1 = Wt + (size_t)wrow1 * wt_stride + 8 * g;
    v8f acc0 = {0.f, 0.f, 0.f, 0.f, 0.f, 0.f, 0.f, 0.f};
    v8f acc1 = {0.f, 0.f, 0.f, 0.f, 0.f, 0.f, 0.f, 0.f};
    for (int k = 0; k < K; k += 32) {
      PackBF a0, b0, b1;
      a0.q.lo = *(const uint4*)(ap0 + k);        // K: k+8g .. k+8g+7
      a0.q.hi = *(const uint4*)(ap0 + k + 16);   // K: k+16+8g .. +7
      b0.q.lo = *(const uint4*)(bp0 + k);        // K: k+8g .. k+8g+7
      b0.q.hi = *(const uint4*)(bp0 + k + 16);   // K: k+16+8g .. +7
      b1.q.lo = *(const uint4*)(bp1 + k);
      b1.q.hi = *(const uint4*)(bp1 + k + 16);
      v16bf av1;
      if constexpr (SHARED_A) {
        av1 = a0.v;
      } else {
        PackBF a1;
        a1.q.lo = *(const uint4*)(ap1 + k);
        a1.q.hi = *(const uint4*)(ap1 + k + 16);
        av1 = a1.v;
      }
      acc0 = __builtin_amdgcn_wmma_f32_16x16x32_bf16(
          false, a0.v, false, b0.v, (short)0, acc0, false, false);
      acc1 = __builtin_amdgcn_wmma_f32_16x16x32_bf16(
          false, av1, false, b1.v, (short)0, acc1, false, false);
    }
    epi_one(acc0, ncol0, bias ? bias[wrow0] : 0.0f);
    epi_one(acc1, ncol1, bias ? bias[wrow1] : 0.0f);
  }
}

// One-time weight convert + transpose into workspace (bf16, [N][K] layout).
__global__ __launch_bounds__(256)
void prep_kernel(const float* __restrict__ Wq, const float* __restrict__ bq,
                 const float* __restrict__ Wk, const float* __restrict__ bk,
                 const float* __restrict__ Wv, const float* __restrict__ bv,
                 const float* __restrict__ Wo,
                 const float* __restrict__ Wm1, const float* __restrict__ Wm2,
                 const float* __restrict__ Ww, const float* __restrict__ bw,
                 const float* __restrict__ Wu, const float* __restrict__ bu,
                 const float* __restrict__ Wkl, const float* __restrict__ Wql,
                 const float* __restrict__ Wq2, const float* __restrict__ Wvl,
                 const float* __restrict__ Wel,
                 unsigned short* __restrict__ wsb,
                 float* __restrict__ bqkv, float* __restrict__ bg)
{
  int id = blockIdx.x * 256 + threadIdx.x;
  if (id >= PREP_ELEMS) return;
  if (id < 786432) {                      // WqkvT[n][k]
    int n = id >> 9, k = id & 511;
    const float* W = (n < 512) ? Wq : ((n < 1024) ? Wk : Wv);
    wsb[OFF_WQKV + id] = f2bf(W[k * 512 + (n & 511)]);
    return;
  }
  id -= 786432;
  if (id < 262144) { int n = id >> 9, k = id & 511; wsb[OFF_WO  + id] = f2bf(Wo [k * 512 + n]); return; }
  id -= 262144;
  if (id < 262144) { int n = id >> 9, k = id & 511; wsb[OFF_WM1 + id] = f2bf(Wm1[k * 512 + n]); return; }
  id -= 262144;
  if (id < 262144) { int n = id >> 9, k = id & 511; wsb[OFF_WM2 + id] = f2bf(Wm2[k * 512 + n]); return; }
  id -= 262144;
  if (id < 1048576) {                     // WgT[n][k] = [Ww;Wu]^T, K=1024
    int n = id >> 10, k = id & 1023;
    float v = (k < 512) ? Ww[k * 1024 + n] : Wu[(k - 512) * 1024 + n];
    wsb[OFF_WG + id] = f2bf(v);
    return;
  }
  id -= 1048576;
  if (id < 20480) {                       // 5 per-head [64][64] transposed
    int m = id >> 12, r = id & 4095, n = r >> 6, k = r & 63;
    const float* W = (m == 0) ? Wkl : (m == 1) ? Wql : (m == 2) ? Wq2 : (m == 3) ? Wvl : Wel;
    wsb[OFF_HEAD + id] = f2bf(W[k * 64 + n]);
    return;
  }
  id -= 20480;
  if (id < 1536) {                        // fused qkv bias
    bqkv[id] = (id < 512) ? bq[id] : (id < 1024) ? bk[id - 512] : bv[id - 1024];
    return;
  }
  id -= 1536;
  bg[id] = bw[id] + bu[id];               // fused gate bias
}

__global__ __launch_bounds__(256)
void relmem_kernel(
    const float* __restrict__ inputs, const float* __restrict__ memory,
    float* __restrict__ out,
    const unsigned short* __restrict__ WqkvT, const unsigned short* __restrict__ WoT,
    const unsigned short* __restrict__ Wm1T, const unsigned short* __restrict__ Wm2T,
    const unsigned short* __restrict__ WgT,
    const unsigned short* __restrict__ WklT, const unsigned short* __restrict__ WqlT,
    const unsigned short* __restrict__ Wq2T, const unsigned short* __restrict__ WvlT,
    const unsigned short* __restrict__ WelT,
    const float* __restrict__ bqkv, const float* __restrict__ bg,
    const float* __restrict__ bo,  const float* __restrict__ bkl,
    const float* __restrict__ bql, const float* __restrict__ bq2,
    const float* __restrict__ bvl, const float* __restrict__ bel,
    const float* __restrict__ bm1, const float* __restrict__ bm2)
{
  extern __shared__ char smem[];
  float*          sM  = (float*)(smem + 0);                  // [16][STR] f32 memory state
  float*          sNM = (float*)(smem + 33280);              // [16][STR] f32 nm
  unsigned short* sKV = (unsigned short*)(smem + 66560);     // [16][STR] bf16 kv
  unsigned short* sAO = (unsigned short*)(smem + 83200);     // bf16 staging (attn out / nm)
  unsigned short* sT1 = (unsigned short*)(smem + 99840);     // bf16 mlp hidden
  unsigned short* sG2 = (unsigned short*)(smem + 116480);    // [16][STRG] bf16 gate A
  unsigned short* sQ  = (unsigned short*)(smem + 149760);
  unsigned short* sK  = (unsigned short*)(smem + 166400);
  unsigned short* sV  = (unsigned short*)(smem + 183040);    // later reused for refined q1
  unsigned short* sQ1 = (unsigned short*)(smem + 199680);
  unsigned short* sQ2 = (unsigned short*)(smem + 216320);
  unsigned short* sKR = (unsigned short*)(smem + 232960);
  unsigned short* sVR = (unsigned short*)(smem + 249600);
  float*          sW  = (float*)(smem + 266240);             // [2][384] qv / scores
  float*          sGT = (float*)(smem + 149760);             // [16][STRG] f32 gates (aliases sQ..sQ1, dead by then)

  const int tid = threadIdx.x;
  const int b0  = blockIdx.x * NB;

  // init memory state (rows b*4+s; s==3 rows zeroed)
  for (int e = tid; e < ROWS * DM; e += 256) {
    int row = e >> 9, col = e & (DM - 1);
    int b = row >> 2, s = row & 3;
    sM[row * STR + col] = (s < 3) ? memory[(size_t)(b0 + b) * (NSLOT * DM) + s * DM + col] : 0.0f;
  }
  __syncthreads();

  for (int t = 0; t < TT; ++t) {
    // ---- kv = [m ; x_t] in bf16 ----
    for (int e = tid; e < ROWS * DM; e += 256) {
      int row = e >> 9, col = e & (DM - 1);
      int b = row >> 2, s = row & 3;
      float v = (s < 3) ? sM[row * STR + col]
                        : inputs[((size_t)(b0 + b) * TT + t) * DM + col];
      sKV[row * STR + col] = f2bf(v);
    }
    // prefetch next timestep's inputs while this step computes
    if (t + 1 < TT) {
      int pb  = tid >> 6;                  // 0..3: batch element
      int off = (tid & 63) << 3;           // 32B-granular coverage of the row
      __builtin_prefetch(&inputs[((size_t)(b0 + pb) * TT + (t + 1)) * DM + off], 0, 1);
    }
    __syncthreads();

    // ---- Q, K, V projections (fused weight, 3 N-slices) ----
    gemm16<true>(sKV, STR, WqkvT,             DM, bqkv,        DM, DM, DM, EPI_BF16, 0, sQ, 0, STR, 0, 0, tid);
    gemm16<true>(sKV, STR, WqkvT + 512 * DM,  DM, bqkv + 512,  DM, DM, DM, EPI_BF16, 0, sK, 0, STR, 0, 0, tid);
    gemm16<true>(sKV, STR, WqkvT + 1024 * DM, DM, bqkv + 1024, DM, DM, DM, EPI_BF16, 0, sV, 0, STR, 0, 0, tid);
    __syncthreads();

    // ---- per-head refinement linears (block-diagonal via wt_mod=64) ----
    gemm16<false>(sK, STR, WklT, DKK, bkl, DM, DKK, DKK, EPI_BF16, 1, sKR, 0, STR, 0, 0, tid);
    gemm16<false>(sQ, STR, WqlT, DKK, bql, DM, DKK, DKK, EPI_BF16, 1, sQ1, 0, STR, 0, 0, tid);
    gemm16<false>(sQ, STR, Wq2T, DKK, bq2, DM, DKK, DKK, EPI_BF16, 1, sQ2, 0, STR, 0, 0, tid);
    gemm16<false>(sV, STR, WvlT, DKK, bvl, DM, DKK, DKK, EPI_BF16, 1, sVR, 0, STR, 0, 0, tid);
    __syncthreads();

    // q1 = relu(q1 @ Wel); raw V dead -> reuse sV as destination
    gemm16<false>(sQ1, STR, WelT, DKK, bel, DM, DKK, DKK, EPI_BF16, 1, sV, 0, STR, 0, 0, tid);
    __syncthreads();

    // ---- tiny einsums: qv = q2.vrf^T, scores = q1.krf^T (len-64 dots) ----
    for (int it = tid; it < 768; it += 256) {
      int pair = it >> 1, which = it & 1;
      int j = pair & 3; int u = pair >> 2;
      int i = u % 3; u /= 3;
      int h = u & 7; int b = u >> 3;
      int rq = (b << 2) + i, rk = (b << 2) + j, cb = h * DKK;
      const unsigned short* pa = which ? (sV  + rq * STR + cb) : (sQ2 + rq * STR + cb);
      const unsigned short* pb = which ? (sKR + rk * STR + cb) : (sVR + rk * STR + cb);
      float acc = 0.0f;
      for (int d = 0; d < DKK; ++d) acc += bf2f(pa[d]) * bf2f(pb[d]);
      sW[which * 384 + pair] = acc;
    }
    __syncthreads();

    // ---- softmax over j, fuse with qv ----
    if (tid < 96) {
      int base = tid * 4;
      float s0 = sW[384 + base], s1 = sW[384 + base + 1];
      float s2 = sW[384 + base + 2], s3 = sW[384 + base + 3];
      float mx = fmaxf(fmaxf(s0, s1), fmaxf(s2, s3));
      float e0 = __expf(s0 - mx), e1 = __expf(s1 - mx);
      float e2 = __expf(s2 - mx), e3 = __expf(s3 - mx);
      float inv = 1.0f / (e0 + e1 + e2 + e3);
      float q0 = sW[base], q1v = sW[base + 1], q2v = sW[base + 2], q3 = sW[base + 3];
      sW[base]     = e0 * inv * q0;
      sW[base + 1] = e1 * inv * q1v;
      sW[base + 2] = e2 * inv * q2v;
      sW[base + 3] = e3 * inv * q3;
    }
    __syncthreads();

    // ---- attn out = (kq*qv) @ K -> bf16 A for Wo ----
    for (int e = tid; e < 12 * DM; e += 256) {
      int col = e & (DM - 1);
      int ri = e >> 9;                       // 0..11 = b*3+i
      int b = ri / 3, i = ri % 3;
      int h = col >> 6;
      int base = (((b * 8 + h) * 3) + i) * 4;
      int rk = b << 2;
      float acc = sW[base + 0] * bf2f(sK[(rk + 0) * STR + col])
                + sW[base + 1] * bf2f(sK[(rk + 1) * STR + col])
                + sW[base + 2] * bf2f(sK[(rk + 2) * STR + col])
                + sW[base + 3] * bf2f(sK[(rk + 3) * STR + col]);
      sAO[((b << 2) + i) * STR + col] = f2bf(acc);
    }
    __syncthreads();

    // ---- nm = m + attn @ Wo + bo ----
    gemm16<true>(sAO, STR, WoT, DM, bo, DM, DM, DM, EPI_NM, 0, 0, sNM, STR, sM, STR, tid);
    __syncthreads();

    // ---- mlp: nm += relu(relu(nm@Wm1+bm1)@Wm2+bm2) ----
    for (int e = tid; e < ROWS * DM; e += 256) {
      int row = e >> 9, col = e & (DM - 1);
      sAO[row * STR + col] = f2bf(sNM[row * STR + col]);
    }
    __syncthreads();
    gemm16<true>(sAO, STR, Wm1T, DM, bm1, DM, DM, DM, EPI_BF16, 1, sT1, 0, STR, 0, 0, tid);
    __syncthreads();
    gemm16<true>(sT1, STR, Wm2T, DM, bm2, DM, DM, DM, EPI_ADDRELU, 1, 0, sNM, STR, 0, 0, tid);
    __syncthreads();

    // ---- gates: A = [x | tanh(m)], gates = A @ [Ww;Wu] + (bw+bu) ----
    for (int e = tid; e < ROWS * 1024; e += 256) {
      int row = e >> 10, col = e & 1023;
      unsigned short v;
      if (col < DM) v = sKV[((row & ~3) + 3) * STR + col];          // x_b (already bf16)
      else          v = f2bf(tanhf(sM[row * STR + (col - DM)]));
      sG2[row * STRG + col] = v;
    }
    __syncthreads();
    gemm16<true>(sG2, STRG, WgT, 1024, bg, 1024, 1024, 1024, EPI_F32, 0, 0, sGT, STRG, 0, 0, tid);
    __syncthreads();

    // ---- m = sigmoid(i)*tanh(nm) + sigmoid(f)*m ; emit output ----
    for (int e = tid; e < 12 * DM; e += 256) {
      int col = e & (DM - 1);
      int ri = e >> 9;
      int b = ri / 3, s = ri % 3;
      int row = (b << 2) + s;
      float ig = sGT[row * STRG + col];
      float fg = sGT[row * STRG + col + DM];
      float nv = sigmoidf_(ig) * tanhf(sNM[row * STR + col])
               + sigmoidf_(fg) * sM[row * STR + col];
      sM[row * STR + col] = nv;
      out[((size_t)(b0 + b) * TT + t) * (NSLOT * DM) + s * DM + col] = nv;
    }
    __syncthreads();
  }
}

extern "C" void kernel_launch(void* const* d_in, const int* in_sizes, int n_in,
                              void* d_out, int out_size, void* d_ws, size_t ws_size,
                              hipStream_t stream) {
  (void)in_sizes; (void)n_in; (void)out_size; (void)ws_size;
  const float* inputs = (const float*)d_in[0];
  const float* memory = (const float*)d_in[1];
  const float* Wq  = (const float*)d_in[2];  const float* bq  = (const float*)d_in[3];
  const float* Wk  = (const float*)d_in[4];  const float* bk  = (const float*)d_in[5];
  const float* Wv  = (const float*)d_in[6];  const float* bv  = (const float*)d_in[7];
  const float* Wo  = (const float*)d_in[8];  const float* bo  = (const float*)d_in[9];
  const float* Wkl = (const float*)d_in[10]; const float* bkl = (const float*)d_in[11];
  const float* Wql = (const float*)d_in[12]; const float* bql = (const float*)d_in[13];
  const float* Wq2 = (const float*)d_in[14]; const float* bq2 = (const float*)d_in[15];
  const float* Wvl = (const float*)d_in[16]; const float* bvl = (const float*)d_in[17];
  const float* Wel = (const float*)d_in[18]; const float* bel = (const float*)d_in[19];
  const float* Wm1 = (const float*)d_in[20]; const float* bm1 = (const float*)d_in[21];
  const float* Wm2 = (const float*)d_in[22]; const float* bm2 = (const float*)d_in[23];
  const float* Ww  = (const float*)d_in[24]; const float* bw  = (const float*)d_in[25];
  const float* Wu  = (const float*)d_in[26]; const float* bu  = (const float*)d_in[27];

  unsigned short* wsb = (unsigned short*)d_ws;
  float* bqkv = (float*)((char*)d_ws + OFF_BQKV_BYTES);
  float* bg   = (float*)((char*)d_ws + OFF_BG_BYTES);

  prep_kernel<<<dim3((PREP_ELEMS + 255) / 256), dim3(256), 0, stream>>>(
      Wq, bq, Wk, bk, Wv, bv, Wo, Wm1, Wm2, Ww, bw, Wu, bu,
      Wkl, Wql, Wq2, Wvl, Wel, wsb, bqkv, bg);

  relmem_kernel<<<dim3(NWG), dim3(256), SMEM_BYTES, stream>>>(
      inputs, memory, (float*)d_out,
      wsb + OFF_WQKV, wsb + OFF_WO, wsb + OFF_WM1, wsb + OFF_WM2, wsb + OFF_WG,
      wsb + OFF_HEAD + 0, wsb + OFF_HEAD + 4096, wsb + OFF_HEAD + 8192,
      wsb + OFF_HEAD + 12288, wsb + OFF_HEAD + 16384,
      bqkv, bg, bo, bkl, bql, bq2, bvl, bel, bm1, bm2);
}
